// SparseConvolution_36481452212697
// MI455X (gfx1250) — matrix operand order repaired and checked by reading, not compiled, not run on hardware
//
#include <hip/hip_runtime.h>

typedef __attribute__((ext_vector_type(16))) _Float16 v16h;
typedef __attribute__((ext_vector_type(8)))  _Float16 v8h;
typedef __attribute__((ext_vector_type(8)))  float    v8f;

#define BATCH 2
#define NPTS  2048
#define CIN   64
#define COUT  128
#define GRID  14          // 12 + 1 pad each side
#define GRID3 (GRID*GRID*GRID)   // 2744
#define NOFF  27
#define KTOT  (NOFF*CIN)  // 1728
#define KSTEPS (KTOT/32)  // 54

// ---------------- ws layout ----------------
// [0, 16384)                     : int keys[B*N]
// [16384, 16384+702464)          : _Float16 voxsumh[B*GRID3*CIN]
// [718848, 718848+442368)        : _Float16 bpack[KSTEPS*8*32*16]
#define WS_KEYS   0
#define WS_VOX    16384
#define WS_BPACK  (16384 + BATCH*GRID3*CIN*2)

// 1) voxel keys on padded 14^3 grid
__global__ void sc_keys(const float* __restrict__ points, int* __restrict__ keys) {
    int id = blockIdx.x * blockDim.x + threadIdx.x;   // [0, B*N)
    if (id >= BATCH * NPTS) return;
    float px = points[id * 3 + 0];
    float py = points[id * 3 + 1];
    float pz = points[id * 3 + 2];
    int vx = (int)px + 1;   // points are positive; trunc == reference int32 cast
    int vy = (int)py + 1;
    int vz = (int)pz + 1;
    keys[id] = (vx * GRID + vy) * GRID + vz;
}

// 2) deterministic per-voxel feature sum -> f16 (ordered scan, no float atomics)
__global__ void sc_voxsum(const float* __restrict__ feat,
                          const int* __restrict__ keys,
                          _Float16* __restrict__ voxsumh) {
    int b = blockIdx.x / GRID3;
    int v = blockIdx.x - b * GRID3;
    int c = threadIdx.x;                 // 64 threads = 2 waves, one channel each
    __shared__ int skeys[NPTS];
    for (int j = threadIdx.x; j < NPTS; j += blockDim.x)
        skeys[j] = keys[b * NPTS + j];
    __syncthreads();
    float acc = 0.0f;
    for (int j = 0; j < NPTS; ++j) {
        if (skeys[j] == v)
            acc += feat[(b * NPTS + j) * CIN + c];
    }
    voxsumh[(b * GRID3 + v) * CIN + c] = (_Float16)acc;
}

// 3) pre-pack weights into exact WMMA B-fragment layout:
//    bpack[ks][ntile][lane][h], h=0..15 -> K = (h&7) + (h>=8 ? 16 : 0) + (lane>=16 ? 8 : 0), N = ntile*16 + lane%16
__global__ void sc_packb(const float* __restrict__ w, _Float16* __restrict__ bp) {
    int id = blockIdx.x * blockDim.x + threadIdx.x;   // KSTEPS*8*32*16 = 221184
    if (id >= KSTEPS * 8 * 32 * 16) return;
    int h    = id & 15;
    int lane = (id >> 4) & 31;
    int nt   = (id >> 9) & 7;
    int ks   = id >> 12;
    int t  = ks >> 1;                         // neighbor index 0..26 (oi*9+oj*3+ok)
    int c  = ((ks & 1) << 5) + (h & 7) + ((h & 8) << 1) + ((lane & 16) ? 8 : 0);  // input channel 0..63
    int o  = (nt << 4) + (lane & 15);         // output channel
    bp[id] = (_Float16)w[(t * CIN + c) * COUT + o];
}

// 4) gathered WMMA GEMM: one wave = one 16x16 output tile, 54 K-steps of 32
__global__ __launch_bounds__(256) void sc_gemm(const int* __restrict__ keys,
                                               const _Float16* __restrict__ voxsumh,
                                               const _Float16* __restrict__ bpack,
                                               const float* __restrict__ bias,
                                               float* __restrict__ out) {
    int tid  = blockIdx.x * blockDim.x + threadIdx.x;
    int wave = tid >> 5;           // [0, 2048): 256 M-tiles x 8 N-tiles
    int lane = tid & 31;
    int ntile = wave & 7;
    int mtile = wave >> 3;         // [0, 256)
    int b  = mtile >> 7;           // batch
    int mt = mtile & 127;          // M-tile within batch

    // A fragment row handled by this lane (M = lane%16), K half (kb) by lane group
    int prow = (mt << 4) + (lane & 15);
    int key  = keys[b * NPTS + prow];
    int kb   = (lane & 16) ? 8 : 0;

    v8f acc = {};
    for (int ks = 0; ks < KSTEPS; ++ks) {
        int t  = ks >> 1;
        int c0 = (ks & 1) << 5;
        int oi = t / 9;
        int r9 = t - oi * 9;
        int oj = r9 / 3;
        int ok = r9 - oj * 3;
        int nkey = key + (oi - 1) * (GRID * GRID) + (oj - 1) * GRID + (ok - 1);

        // A: two b128 loads of contiguous channels (16-bit A 16x32 layout)
        const _Float16* ap = voxsumh + ((b * GRID3 + nkey) << 6) + c0 + kb;
        v8h a0 = *(const v8h*)(ap);        // K = kb .. kb+7
        v8h a1 = *(const v8h*)(ap + 16);   // K = 16+kb .. 16+kb+7
        v16h A = __builtin_shufflevector(a0, a1, 0,1,2,3,4,5,6,7,8,9,10,11,12,13,14,15);

        // B: two b128 loads, perfectly coalesced (pre-packed fragment)
        const _Float16* bp = bpack + (((ks << 3) + ntile) << 9) + (lane << 4);
        v8h b0 = *(const v8h*)(bp);
        v8h b1 = *(const v8h*)(bp + 8);
        v16h Bm = __builtin_shufflevector(b0, b1, 0,1,2,3,4,5,6,7,8,9,10,11,12,13,14,15);

        acc = __builtin_amdgcn_wmma_f32_16x16x32_f16(
            /*neg_a=*/false, A, /*neg_b=*/false, Bm,
            /*c_mod=*/(short)0, acc, /*reuse_a=*/false, /*reuse_b=*/false);
    }

    // C/D layout: VGPR r -> M = r + (lane>=16 ? 8 : 0), N = lane%16
    int col   = (ntile << 4) + (lane & 15);
    float bv  = bias[col];
    int rbase = (mt << 4) + ((lane & 16) ? 8 : 0);
    #pragma unroll
    for (int r = 0; r < 8; ++r)
        out[(b * NPTS + rbase + r) * COUT + col] = acc[r] + bv;
}

extern "C" void kernel_launch(void* const* d_in, const int* in_sizes, int n_in,
                              void* d_out, int out_size, void* d_ws, size_t ws_size,
                              hipStream_t stream) {
    const float* points = (const float*)d_in[0];   // (B,N,3)  f32
    const float* feats  = (const float*)d_in[1];   // (B,N,64) f32
    const float* weight = (const float*)d_in[2];   // (3,3,3,64,128) f32
    const float* bias   = (const float*)d_in[3];   // (128,)   f32
    float* out = (float*)d_out;                    // (B,N,128) f32

    char* ws = (char*)d_ws;
    int*      keys    = (int*)(ws + WS_KEYS);
    _Float16* voxsumh = (_Float16*)(ws + WS_VOX);
    _Float16* bpack   = (_Float16*)(ws + WS_BPACK);

    // 1) voxel keys
    sc_keys<<<(BATCH * NPTS + 255) / 256, 256, 0, stream>>>(points, keys);
    // 2) per-voxel sums (all 14^3 voxels written -> no zero-init needed)
    sc_voxsum<<<BATCH * GRID3, 64, 0, stream>>>(feats, keys, voxsumh);
    // 3) pack weights into B fragments
    sc_packb<<<(KSTEPS * 8 * 32 * 16 + 255) / 256, 256, 0, stream>>>(weight, bpack);
    // 4) WMMA GEMM: 2048 waves = 256 blocks x 8 waves
    sc_gemm<<<256, 256, 0, stream>>>(keys, voxsumh, bpack, bias, out);
}
